// Block_67585605370049
// MI455X (gfx1250) — compile-verified, matrix operand order checked by reading, not run on hardware
//
#include <hip/hip_runtime.h>
#include <hip/hip_bf16.h>

typedef __bf16 bf16;
typedef __attribute__((ext_vector_type(16))) __bf16 bf16x16;
typedef __attribute__((ext_vector_type(8)))  float  floatx8;
typedef __attribute__((ext_vector_type(4)))  unsigned int uint32x4;
typedef __attribute__((ext_vector_type(8)))  unsigned int uint32x8;

// ---------------------------------------------------------------------------
// WMMA wrapper: D = A(16x32 bf16) * B(32x16 bf16) + C(16x16 f32)
// ---------------------------------------------------------------------------
__device__ __forceinline__ floatx8 wmma_bf16(bf16x16 a, bf16x16 b, floatx8 c) {
    return __builtin_amdgcn_wmma_f32_16x16x32_bf16(
        /*neg_a=*/false, a, /*neg_b=*/false, b,
        /*c_mod=*/(short)0, c, /*reuse_a=*/false, /*reuse_b=*/false);
}

__device__ __forceinline__ floatx8 fzero8() {
    floatx8 z;
#pragma unroll
    for (int i = 0; i < 8; ++i) z[i] = 0.0f;
    return z;
}

// ---------------------------------------------------------------------------
// gfx1250 async memory->LDS copy (no VGPR round trip, tracked by ASYNCcnt).
// LDS address = low 32 bits of a generic pointer to __shared__ memory.
// ---------------------------------------------------------------------------
__device__ __forceinline__ void async_load_b128(const bf16* gptr, bf16* lptr) {
    unsigned lds = (unsigned)(size_t)lptr;
    unsigned long long ga = (unsigned long long)(size_t)gptr;
    asm volatile("global_load_async_to_lds_b128 %0, %1, off"
                 :: "v"(lds), "v"(ga)
                 : "memory");
}
__device__ __forceinline__ void wait_async0() {
    asm volatile("s_wait_asynccnt 0x0" ::: "memory");
}

// ---------------------------------------------------------------------------
// gfx1250 Tensor Data Mover: 2D tile (rows x cols bf16 elements, row pitch
// `stride` elements) from global memory into LDS, with hardware LDS padding
// of 4 DWORDs after every 32 DWORDs (i.e. 64-elem rows land in 72-elem rows).
// Descriptor per ISA sec.8: group0 = {count, lds_addr, global_addr, type=2},
// group1 = {data_size=2B, pad cfg, tensor/tile dims, dim0 stride}.
// Issued once per wave (EXEC ignored); completion tracked by TENSORcnt.
// ---------------------------------------------------------------------------
__device__ __forceinline__ void tdm_load_2d_pad(const bf16* gptr, bf16* lptr,
                                                unsigned rows, unsigned cols,
                                                unsigned stride_elems) {
    unsigned long long ga = (unsigned long long)(size_t)gptr;
    uint32x4 g0;
    g0[0] = 1u;                                   // count=1, user descriptor
    g0[1] = (unsigned)(size_t)lptr;               // lds_addr (bytes)
    g0[2] = (unsigned)ga;                         // global_addr[31:0]
    g0[3] = (unsigned)(ga >> 32) | 0x80000000u;   // global_addr[56:32] | type=2
    uint32x8 g1;
    // data_size=1 (2 bytes), pad_enable, pad_interval=4 (32 DW), pad_amount=3 (4 DW)
    g1[0] = (1u << 16) | (1u << 20) | (4u << 22) | (3u << 25);
    g1[1] = (cols & 0xffffu) << 16;               // tensor_dim0[15:0]
    g1[2] = (rows & 0xffffu) << 16;               // tensor_dim0[31:16]=0, tensor_dim1 lo
    g1[3] = (cols & 0xffffu) << 16;               // tensor_dim1 hi=0, tile_dim0
    g1[4] = (rows & 0xffffu);                     // tile_dim1, tile_dim2=0
    g1[5] = stride_elems;                         // tensor_dim0_stride[31:0]
    g1[6] = 0u;                                   // dim0_stride hi, dim1_stride lo
    g1[7] = 0u;                                   // dim1_stride hi
    asm volatile("tensor_load_to_lds %0, %1" :: "s"(g0), "s"(g1) : "memory");
}
__device__ __forceinline__ void wait_tensor0() {
#if __has_builtin(__builtin_amdgcn_s_wait_tensorcnt)
    __builtin_amdgcn_s_wait_tensorcnt(0);
#else
    asm volatile("s_wait_tensorcnt 0x0" ::: "memory");
#endif
}

// Build a 16x32 bf16 fragment from a row-major LDS tile.
// p points at element [row][c0] where c0 = 8*(lane>=16).
// Lane layout (ISA 7.12.2, 16-bit A 16x32): lane<16 holds K {0..7,16..23},
// lane>=16 holds K {8..15,24..31}; both are two contiguous 8-element runs.
__device__ __forceinline__ bf16x16 frag_row(const bf16* p) {
    bf16x16 f;
#pragma unroll
    for (int e = 0; e < 8; ++e) f[e] = p[e];
#pragma unroll
    for (int e = 0; e < 8; ++e) f[e + 8] = p[e + 16];
    return f;
}

// Fragment from a column of a row-major [K][N] LDS tile (used for V in P*V).
__device__ __forceinline__ bf16x16 frag_col(const bf16* base, int ld, int col, int c0) {
    bf16x16 f;
#pragma unroll
    for (int e = 0; e < 8; ++e) f[e] = base[(c0 + e) * ld + col];
#pragma unroll
    for (int e = 0; e < 8; ++e) f[e + 8] = base[(c0 + 16 + e) * ld + col];
    return f;
}

// wave32 full reduction
__device__ __forceinline__ float wave_sum32(float v) {
#pragma unroll
    for (int m = 16; m >= 1; m >>= 1) v += __shfl_xor(v, m, 32);
    return v;
}
// reduction within each 16-lane half (rows of a 16x16 WMMA D tile)
__device__ __forceinline__ float red16_max(float v) {
#pragma unroll
    for (int m = 8; m >= 1; m >>= 1) v = fmaxf(v, __shfl_xor(v, m, 32));
    return v;
}
__device__ __forceinline__ float red16_sum(float v) {
#pragma unroll
    for (int m = 8; m >= 1; m >>= 1) v += __shfl_xor(v, m, 32);
    return v;
}

// ---------------------------------------------------------------------------
// f32 -> bf16 conversion (vectorized float4 loads)
// ---------------------------------------------------------------------------
__global__ void cvt_bf16_kernel(const float* __restrict__ in, bf16* __restrict__ out, long n) {
    long i = ((long)blockIdx.x * blockDim.x + threadIdx.x) * 4;
    if (i < n) {
        float4 v = *(const float4*)(in + i);
        out[i + 0] = (bf16)v.x;
        out[i + 1] = (bf16)v.y;
        out[i + 2] = (bf16)v.z;
        out[i + 3] = (bf16)v.w;
    }
}

// ---------------------------------------------------------------------------
// LayerNorm: one block per row, f32 in, bf16 out
// ---------------------------------------------------------------------------
__global__ __launch_bounds__(256) void ln_bf16_kernel(
    const float* __restrict__ x, const float* __restrict__ g,
    const float* __restrict__ bta, bf16* __restrict__ out, int C) {
    const int row = blockIdx.x;
    const float* xr = x + (size_t)row * C;
    float ls = 0.f, lq = 0.f;
    for (int i = threadIdx.x; i < C; i += 256) {
        float v = xr[i];
        ls += v;
        lq += v * v;
    }
    ls = wave_sum32(ls);
    lq = wave_sum32(lq);
    __shared__ float ss[8], sq[8];
    int lane = threadIdx.x & 31, w = threadIdx.x >> 5;
    if (lane == 0) { ss[w] = ls; sq[w] = lq; }
    __syncthreads();
    if (w == 0) {
        float a  = (lane < 8) ? ss[lane] : 0.f;
        float b2 = (lane < 8) ? sq[lane] : 0.f;
        a = wave_sum32(a);
        b2 = wave_sum32(b2);
        if (lane == 0) { ss[0] = a; sq[0] = b2; }
    }
    __syncthreads();
    float mean = ss[0] / (float)C;
    float var  = sq[0] / (float)C - mean * mean;
    float inv  = rsqrtf(var + 1e-5f);
    bf16* orow = out + (size_t)row * C;
    for (int i = threadIdx.x; i < C; i += 256)
        orow[i] = (bf16)((xr[i] - mean) * inv * g[i] + bta[i]);
}

// ---------------------------------------------------------------------------
// GEMM: C[M,N] = A[M,K](bf16) * B[K,N](bf16) + bias, with fused epilogues.
// Block = 256 threads (8 waves), tile 128x128, K-step 32, double-buffered:
// async global->LDS loads for tile t+1 are issued before the WMMAs of tile t
// and waited only afterwards (ASYNCcnt hides HBM/L2 latency behind compute).
// EPI: 0 = bf16 out, 1 = bf16 out + tanh-GELU, 2 = f32 out + f32 residual add.
// ---------------------------------------------------------------------------
template <int EPI>
__global__ __launch_bounds__(256) void gemm_bf16_kernel(
    const bf16* __restrict__ A, const bf16* __restrict__ B,
    const float* __restrict__ bias, const float* __restrict__ res,
    void* __restrict__ out, int M, int N, int K) {
    __shared__ bf16 As[2][128][40];  // row-major [m][k], 80B rows (16B aligned)
    __shared__ bf16 Bs[2][128][40];  // transposed [n][k]

    const int tid  = threadIdx.x;
    const int lane = tid & 31;
    const int wave = tid >> 5;
    const int wr = wave >> 1;   // 0..3 : 32-row group
    const int wc = wave & 1;    // 0..1 : 64-col group
    const int hf = lane >> 4;   // lane half
    const int c0 = hf << 3;
    const int ln = lane & 15;
    const int bm = blockIdx.y * 128;
    const int bn = blockIdx.x * 128;

    floatx8 acc[2][4];
#pragma unroll
    for (int i = 0; i < 2; ++i)
#pragma unroll
        for (int j = 0; j < 4; ++j) acc[i][j] = fzero8();

    const int ar = tid >> 1, ah = tid & 1;  // A: 128 rows x 2 half-rows of 16
    const int br = tid >> 3, bsg = tid & 7; // B: 32 rows x 8 segs of 16

    // ---- prologue: stage tile 0 into buffer 0 ----
    {
        const bf16* ga = A + (size_t)(bm + ar) * K + ah * 16;
        async_load_b128(ga + 0, &As[0][ar][ah * 16 + 0]);
        async_load_b128(ga + 8, &As[0][ar][ah * 16 + 8]);
        const bf16* gb = B + (size_t)br * N + bn + bsg * 16;
        alignas(16) bf16 t0[16];
        *(uint4*)&t0[0] = ((const uint4*)gb)[0];
        *(uint4*)&t0[8] = ((const uint4*)gb)[1];
#pragma unroll
        for (int j = 0; j < 16; ++j) Bs[0][bsg * 16 + j][br] = t0[j];
        wait_async0();
    }
    __syncthreads();

    int cur = 0;
    for (int kk = 0; kk < K; kk += 32) {
        const int nxt = cur ^ 1;
        const bool has_next = (kk + 32 < K);
        alignas(16) bf16 breg[16];
        if (has_next) {
            // issue next tile: async A -> LDS[nxt], B -> VGPRs (both overlap WMMAs)
            const bf16* ga = A + (size_t)(bm + ar) * K + (kk + 32) + ah * 16;
            async_load_b128(ga + 0, &As[nxt][ar][ah * 16 + 0]);
            async_load_b128(ga + 8, &As[nxt][ar][ah * 16 + 8]);
            const bf16* gb = B + (size_t)(kk + 32 + br) * N + bn + bsg * 16;
            *(uint4*)&breg[0] = ((const uint4*)gb)[0];
            *(uint4*)&breg[8] = ((const uint4*)gb)[1];
            if (kk + 64 < K) {  // L2 warm-up two tiles ahead (global_prefetch_b8)
                __builtin_prefetch(A + (size_t)(bm + ar) * K + kk + 64 + ah * 16, 0, 1);
                __builtin_prefetch(B + (size_t)(kk + 64 + br) * N + bn + bsg * 16, 0, 1);
            }
        }

        // ---- compute current tile: 8 WMMAs per wave ----
        bf16x16 af[2], bfv[4];
#pragma unroll
        for (int mt = 0; mt < 2; ++mt)
            af[mt] = frag_row(&As[cur][wr * 32 + mt * 16 + ln][c0]);
#pragma unroll
        for (int nt = 0; nt < 4; ++nt)
            bfv[nt] = frag_row(&Bs[cur][wc * 64 + nt * 16 + ln][c0]);
#pragma unroll
        for (int mt = 0; mt < 2; ++mt)
#pragma unroll
            for (int nt = 0; nt < 4; ++nt)
                acc[mt][nt] = wmma_bf16(af[mt], bfv[nt], acc[mt][nt]);

        if (has_next) {
            // commit staged B to LDS[nxt], then close the async pipeline stage
#pragma unroll
            for (int j = 0; j < 16; ++j) Bs[nxt][bsg * 16 + j][br] = breg[j];
            wait_async0();
            __syncthreads();
        }
        cur = nxt;
    }

    // ---- epilogue: D layout = lane holds col n=ln, rows v + 8*hf ----
#pragma unroll
    for (int mt = 0; mt < 2; ++mt) {
#pragma unroll
        for (int nt = 0; nt < 4; ++nt) {
            const int n  = bn + wc * 64 + nt * 16 + ln;
            const float bs = bias[n];
#pragma unroll
            for (int v = 0; v < 8; ++v) {
                const int m = bm + wr * 32 + mt * 16 + v + 8 * hf;
                float val = acc[mt][nt][v] + bs;
                const size_t idx = (size_t)m * N + n;
                if (EPI == 0) {
                    ((bf16*)out)[idx] = (bf16)val;
                } else if (EPI == 1) {
                    float t = 0.7978845608028654f * (val + 0.044715f * val * val * val);
                    val = 0.5f * val * (1.0f + tanhf(t));
                    ((bf16*)out)[idx] = (bf16)val;
                } else {
                    ((float*)out)[idx] = val + res[idx];
                }
            }
        }
    }
}

// ---------------------------------------------------------------------------
// Fused flash-style attention (shared by causal self-attn and cross-attn).
// Block = 128 threads (4 waves); each wave owns 16 query rows; 64 q rows/block.
// K/V chunks are staged by the Tensor Data Mover: wave 0 issues two 2D-tile
// descriptors (32 x 128B, hardware-padded into 72-elem LDS rows), waits on
// TENSORcnt, and the workgroup barrier publishes the LDS. dh = 64.
// ---------------------------------------------------------------------------
__global__ __launch_bounds__(128) void attn_kernel(
    const bf16* __restrict__ Q, int qstride,
    const bf16* __restrict__ Kp, const bf16* __restrict__ Vp, int kvstride,
    bf16* __restrict__ O, int Tq, int Tk, int causal, float scale) {
    __shared__ bf16 Ks[32][72];        // K chunk [key][d]  (144B rows, 16B aligned)
    __shared__ bf16 Vs[32][72];        // V chunk [key][d]
    __shared__ bf16 Ps[4][16][40];     // per-wave P staging [q][key]

    const int tid  = threadIdx.x;
    const int lane = tid & 31;
    const int w    = tid >> 5;
    const int hf = lane >> 4, c0 = hf << 3, ln = lane & 15;
    const int q0 = blockIdx.x * 64;
    const int h  = blockIdx.y;
    const int b  = blockIdx.z;

    const bf16* qbase = Q  + (size_t)(b * Tq) * qstride  + h * 64;
    const bf16* kbase = Kp + (size_t)(b * Tk) * kvstride + h * 64;
    const bf16* vbase = Vp + (size_t)(b * Tk) * kvstride + h * 64;

    // Q fragments for this wave's 16 rows (2 K-steps over dh=64), kept resident
    bf16x16 qf[2];
    {
        const bf16* qp = qbase + (size_t)(q0 + w * 16 + ln) * qstride;
#pragma unroll
        for (int ks = 0; ks < 2; ++ks) {
            bf16x16 f;
#pragma unroll
            for (int e = 0; e < 8; ++e) f[e] = qp[ks * 32 + c0 + e];
#pragma unroll
            for (int e = 0; e < 8; ++e) f[e + 8] = qp[ks * 32 + c0 + 16 + e];
            qf[ks] = f;
        }
    }

    floatx8 o[4];
#pragma unroll
    for (int i = 0; i < 4; ++i) o[i] = fzero8();
    float m_r[8], l_r[8];
#pragma unroll
    for (int v = 0; v < 8; ++v) { m_r[v] = -1e30f; l_r[v] = 0.0f; }

    int kend = causal ? (q0 + 64) : Tk;
    if (kend > Tk) kend = Tk;

    for (int kc = 0; kc < kend; kc += 32) {
        __syncthreads();
        if (w == 0) {
            // TDM: one descriptor per tile, issued once by wave 0 (EXEC ignored)
            tdm_load_2d_pad(kbase + (size_t)kc * kvstride, &Ks[0][0], 32, 64, kvstride);
            tdm_load_2d_pad(vbase + (size_t)kc * kvstride, &Vs[0][0], 32, 64, kvstride);
            wait_tensor0();
        }
        __syncthreads();

        // S = Q * K^T : two 16x16 n-tiles, K-dim = dh = 64 (2 WMMA steps)
        floatx8 s[2];
#pragma unroll
        for (int nt = 0; nt < 2; ++nt) {
            floatx8 a = fzero8();
#pragma unroll
            for (int ks = 0; ks < 2; ++ks) {
                bf16x16 kf = frag_row(&Ks[nt * 16 + ln][ks * 32 + c0]);
                a = wmma_bf16(qf[ks], kf, a);
            }
            s[nt] = a;
        }
        // scale + causal mask (D layout: lane holds col n=ln, rows v+8*hf)
#pragma unroll
        for (int nt = 0; nt < 2; ++nt) {
            const int kg = kc + nt * 16 + ln;
#pragma unroll
            for (int v = 0; v < 8; ++v) {
                float x = s[nt][v] * scale;
                if (causal) {
                    const int qg = q0 + w * 16 + v + 8 * hf;
                    if (kg > qg) x = -1e30f;
                }
                s[nt][v] = x;
            }
        }
        // online softmax
        float pr0[8], pr1[8], alpha[8];
#pragma unroll
        for (int v = 0; v < 8; ++v) {
            float t = fmaxf(s[0][v], s[1][v]);
            t = red16_max(t);
            float mn = fmaxf(m_r[v], t);
            alpha[v] = __expf(m_r[v] - mn);
            m_r[v]   = mn;
            pr0[v] = __expf(s[0][v] - mn);
            pr1[v] = __expf(s[1][v] - mn);
            float rs = red16_sum(pr0[v] + pr1[v]);
            l_r[v] = l_r[v] * alpha[v] + rs;
        }
#pragma unroll
        for (int nt = 0; nt < 4; ++nt)
#pragma unroll
            for (int v = 0; v < 8; ++v) o[nt][v] *= alpha[v];

        // D-layout -> A-layout for P via per-wave LDS (same-wave LDS is in-order)
#pragma unroll
        for (int v = 0; v < 8; ++v) {
            Ps[w][v + 8 * hf][ln]      = (bf16)pr0[v];
            Ps[w][v + 8 * hf][16 + ln] = (bf16)pr1[v];
        }
        bf16x16 pf = frag_row(&Ps[w][ln][c0]);
        // O += P(16x32) * V(32x64)
#pragma unroll
        for (int nt = 0; nt < 4; ++nt) {
            bf16x16 vf = frag_col(&Vs[0][0], 72, nt * 16 + ln, c0);
            o[nt] = wmma_bf16(pf, vf, o[nt]);
        }
    }

    // finalize: divide by l, write bf16 rows (lanes 0..15 write 16 contiguous cols)
#pragma unroll
    for (int v = 0; v < 8; ++v) {
        const float inv = 1.0f / l_r[v];
        const int qg = q0 + w * 16 + v + 8 * hf;
        bf16* op = O + (size_t)(b * Tq + qg) * 1024 + h * 64;
#pragma unroll
        for (int nt = 0; nt < 4; ++nt)
            op[nt * 16 + ln] = (bf16)(o[nt][v] * inv);
    }
}

// ---------------------------------------------------------------------------
// Host-side orchestration
// ---------------------------------------------------------------------------
static void cvt(const float* in, bf16* out, long n, hipStream_t s) {
    long n4 = n / 4;
    cvt_bf16_kernel<<<dim3((unsigned)((n4 + 255) / 256)), 256, 0, s>>>(in, out, n);
}

template <int EPI>
static void gemm(const bf16* A, const bf16* B, const float* bias, const float* res,
                 void* out, int M, int N, int K, hipStream_t s) {
    dim3 g(N / 128, M / 128);
    gemm_bf16_kernel<EPI><<<g, 256, 0, s>>>(A, B, bias, res, out, M, N, K);
}

extern "C" void kernel_launch(void* const* d_in, const int* in_sizes, int n_in,
                              void* d_out, int out_size, void* d_ws, size_t ws_size,
                              hipStream_t stream) {
    const float* x      = (const float*)d_in[0];
    const float* key    = (const float*)d_in[1];
    const float* val    = (const float*)d_in[2];
    const float* ln1_g  = (const float*)d_in[3];
    const float* ln1_b  = (const float*)d_in[4];
    const float* ln2_g  = (const float*)d_in[5];
    const float* ln2_b  = (const float*)d_in[6];
    const float* ln3_g  = (const float*)d_in[7];
    const float* ln3_b  = (const float*)d_in[8];
    const float* W_attn = (const float*)d_in[9];
    const float* b_attn = (const float*)d_in[10];
    const float* W_aproj= (const float*)d_in[11];
    const float* b_aproj= (const float*)d_in[12];
    const float* Wq     = (const float*)d_in[13];
    const float* bq     = (const float*)d_in[14];
    const float* Wk     = (const float*)d_in[15];
    const float* bk     = (const float*)d_in[16];
    const float* Wv     = (const float*)d_in[17];
    const float* bv     = (const float*)d_in[18];
    const float* W_cproj= (const float*)d_in[19];
    const float* b_cproj= (const float*)d_in[20];
    const float* W_fc   = (const float*)d_in[21];
    const float* b_fc   = (const float*)d_in[22];
    const float* W_mproj= (const float*)d_in[23];
    const float* b_mproj= (const float*)d_in[24];

    const int B = 8, T = 1024, C = 1024, Ck = 768, TK = 256, NH = 16;
    const size_t BT  = (size_t)B * T;    // 8192 rows
    const size_t BTk = (size_t)B * TK;   // 2048 rows

    char* ws = (char*)d_ws;
    size_t off = 0;
    auto alloc = [&](size_t bytes) -> void* {
        void* p = ws + off;
        off += (bytes + 255) & ~(size_t)255;
        return p;
    };

    // persistent bf16 buffers
    bf16* wattn  = (bf16*)alloc((size_t)C * 3 * C * 2);
    bf16* waproj = (bf16*)alloc((size_t)C * C * 2);
    bf16* wqb    = (bf16*)alloc((size_t)C * C * 2);
    bf16* wkb    = (bf16*)alloc((size_t)Ck * C * 2);
    bf16* wvb    = (bf16*)alloc((size_t)Ck * C * 2);
    bf16* wcproj = (bf16*)alloc((size_t)C * C * 2);
    bf16* wfc    = (bf16*)alloc((size_t)C * 4 * C * 2);
    bf16* wmproj = (bf16*)alloc((size_t)4 * C * C * 2);
    bf16* keyb   = (bf16*)alloc(BTk * Ck * 2);
    bf16* valb   = (bf16*)alloc(BTk * Ck * 2);
    bf16* lnbuf  = (bf16*)alloc(BT * C * 2);
    bf16* attnb  = (bf16*)alloc(BT * C * 2);
    // phase-aliased scratch (max of: qkv 50MB | q/k/v-proj 24MB | mlp hidden 67MB)
    char* scratch = (char*)alloc(BT * (size_t)4 * C * 2);
    bf16* qkv = (bf16*)scratch;
    bf16* qc  = (bf16*)scratch;
    bf16* kc  = (bf16*)(scratch + BT * C * 2);
    bf16* vc  = (bf16*)(scratch + BT * C * 2 + BTk * C * 2);
    bf16* hfc = (bf16*)scratch;

    float* xo = (float*)d_out;  // running residual state lives in d_out x-region

    // 0) one-time down-conversions
    cvt(W_attn, wattn, (long)C * 3 * C, stream);
    cvt(W_aproj, waproj, (long)C * C, stream);
    cvt(Wq, wqb, (long)C * C, stream);
    cvt(Wk, wkb, (long)Ck * C, stream);
    cvt(Wv, wvb, (long)Ck * C, stream);
    cvt(W_cproj, wcproj, (long)C * C, stream);
    cvt(W_fc, wfc, (long)C * 4 * C, stream);
    cvt(W_mproj, wmproj, (long)4 * C * C, stream);
    cvt(key, keyb, (long)BTk * Ck, stream);
    cvt(val, valb, (long)BTk * Ck, stream);

    const float scale = 0.125f;  // 1/sqrt(64)

    // 1) self-attention branch
    ln_bf16_kernel<<<dim3((unsigned)BT), 256, 0, stream>>>(x, ln1_g, ln1_b, lnbuf, C);
    gemm<0>(lnbuf, wattn, b_attn, nullptr, qkv, (int)BT, 3 * C, C, stream);
    attn_kernel<<<dim3(T / 64, NH, B), 128, 0, stream>>>(
        qkv, 3 * C, qkv + C, qkv + 2 * C, 3 * C, attnb, T, T, /*causal=*/1, scale);
    gemm<2>(attnb, waproj, b_aproj, x, xo, (int)BT, C, C, stream);  // x + attn

    // 2) cross-attention branch
    ln_bf16_kernel<<<dim3((unsigned)BT), 256, 0, stream>>>(xo, ln3_g, ln3_b, lnbuf, C);
    gemm<0>(lnbuf, wqb, bq, nullptr, qc, (int)BT, C, C, stream);
    gemm<0>(keyb, wkb, bk, nullptr, kc, (int)BTk, C, Ck, stream);
    gemm<0>(valb, wvb, bv, nullptr, vc, (int)BTk, C, Ck, stream);
    attn_kernel<<<dim3(T / 64, NH, B), 128, 0, stream>>>(
        qc, C, kc, vc, C, attnb, T, TK, /*causal=*/0, scale);
    gemm<2>(attnb, wcproj, b_cproj, xo, xo, (int)BT, C, C, stream);  // x += cross

    // 3) MLP branch
    ln_bf16_kernel<<<dim3((unsigned)BT), 256, 0, stream>>>(xo, ln2_g, ln2_b, lnbuf, C);
    gemm<1>(lnbuf, wfc, b_fc, nullptr, hfc, (int)BT, 4 * C, C, stream);   // + GELU
    gemm<2>(hfc, wmproj, b_mproj, xo, xo, (int)BT, C, 4 * C, stream);     // x += mlp

    // 4) echo key/val into the output tuple
    size_t x_elems = BT * (size_t)C;
    size_t kv_elems = BTk * (size_t)Ck;
    hipMemcpyAsync(xo + x_elems, key, kv_elems * sizeof(float),
                   hipMemcpyDeviceToDevice, stream);
    hipMemcpyAsync(xo + x_elems + kv_elems, val, kv_elems * sizeof(float),
                   hipMemcpyDeviceToDevice, stream);
}
